// SelfAttentionLayer_12103217840334
// MI455X (gfx1250) — matrix operand order repaired and checked by reading, not compiled, not run on hardware
//
#include <hip/hip_runtime.h>
#include <hip/hip_bf16.h>
#include <math.h>

typedef __attribute__((ext_vector_type(16))) _Float16 v16h;
typedef __attribute__((ext_vector_type(8)))  float    v8f;

#define N_  1024
#define C_  256
#define H_  8
#define DH_ 32

// ---------------------------------------------------------------------------
// Kernel 1: convert projection weights to f16. Weights are (C,C) row-major
// (out, in) which is exactly the "B^T" (col-major-B / K-contiguous) layout
// WMMA wants.
__global__ void wconv_kernel(const float* __restrict__ Wq,
                             const float* __restrict__ Wk,
                             const float* __restrict__ Wv,
                             _Float16* __restrict__ Wqh,
                             _Float16* __restrict__ Wkh,
                             _Float16* __restrict__ Wvh) {
    int idx = blockIdx.x * blockDim.x + threadIdx.x;
    if (idx < C_ * C_) {
        Wqh[idx] = (_Float16)Wq[idx];
        Wkh[idx] = (_Float16)Wk[idx];
        Wvh[idx] = (_Float16)Wv[idx];
    }
}

// ---------------------------------------------------------------------------
// Kernel 2: fused SE module + query_pos add; emits qk (f16) and tgt (f16).
__global__ void se_qk_kernel(const float* __restrict__ tgt,
                             const float* __restrict__ qpos,
                             const float* __restrict__ qedge,
                             const float* __restrict__ se_w1,   // (32,2)
                             const float* __restrict__ se_w2,   // (2,32)
                             _Float16* __restrict__ qkh,
                             _Float16* __restrict__ tgth) {
    int idx = blockIdx.x * blockDim.x + threadIdx.x;
    if (idx >= N_ * C_) return;
    float reg  = tgt[idx];
    float edge = qedge[idx];
    float a0 = 0.f, a1 = 0.f;
#pragma unroll
    for (int j = 0; j < 32; ++j) {
        float h = se_w1[2 * j] * reg + se_w1[2 * j + 1] * edge;
        h = fmaxf(h, 0.f);
        a0 += se_w2[j] * h;
        a1 += se_w2[32 + j] * h;
    }
    float w0 = 1.f / (1.f + __expf(-a0));
    float w1 = 1.f / (1.f + __expf(-a1));
    float qk = reg * w0 + edge * w1 + qpos[idx];
    qkh[idx]  = (_Float16)qk;
    tgth[idx] = (_Float16)reg;
}

// ---------------------------------------------------------------------------
// WMMA fragment loaders (contiguous-k order k = 16*(lane>>4)+i, wave32).
__device__ inline v16h load_a_frag(const _Float16* base, int ldk, int lane) {
    // 16x32 f16 A: row = lane&15; K = 16*(lane>>4) .. +15 contiguous.
    int ln = lane & 15, hi = lane >> 4;
    const _Float16* p = base + ln * ldk + hi * 16;
    v16h a;
#pragma unroll
    for (int i = 0; i < 16; ++i) a[i] = p[i];
    return a;
}
__device__ inline v16h load_b_frag(const _Float16* baseT, int ldk, int lane) {
    // 32x16 f16 B given as B^T (row = output col, K contiguous):
    // lane holds col = lane&15, K = 16*(lane>>4) .. +15 contiguous.
    int ln = lane & 15, hi = lane >> 4;
    const _Float16* p = baseT + ln * ldk + hi * 16;
    v16h b;
#pragma unroll
    for (int i = 0; i < 16; ++i) b[i] = p[i];
    return b;
}

// ---------------------------------------------------------------------------
// Kernel 3: projections Y = X @ W^T  (one wave = one 16x16 tile; z picks Q/K/V)
__global__ void proj_kernel(const _Float16* __restrict__ qkh,
                            const _Float16* __restrict__ tgth,
                            const _Float16* __restrict__ Wqh,
                            const _Float16* __restrict__ Wkh,
                            const _Float16* __restrict__ Wvh,
                            float* __restrict__ Q,
                            float* __restrict__ K,
                            float* __restrict__ V) {
    int z = blockIdx.z;
    const _Float16* X = (z == 2) ? tgth : qkh;
    const _Float16* W = (z == 0) ? Wqh : (z == 1) ? Wkh : Wvh;
    float* Y = (z == 0) ? Q : (z == 1) ? K : V;
    int lane = threadIdx.x & 31;
    int ln = lane & 15, hi = lane >> 4;
    int nb = blockIdx.x * 16, ob = blockIdx.y * 16;
    v8f acc = {};
    for (int k0 = 0; k0 < C_; k0 += 32) {
        v16h a = load_a_frag(X + nb * C_ + k0, C_, lane);
        v16h b = load_b_frag(W + ob * C_ + k0, C_, lane);
        acc = __builtin_amdgcn_wmma_f32_16x16x32_f16(false, a, false, b,
                                                     (short)0, acc, false, false);
    }
#pragma unroll
    for (int r = 0; r < 8; ++r)
        Y[(nb + r + hi * 8) * C_ + ob + ln] = acc[r];
}

// ---------------------------------------------------------------------------
// Kernel 4: per-(n,h) L2 normalize of q,k (32 lanes = 32 head dims), cast f16.
__global__ void norm_kernel(const float* __restrict__ Q,
                            const float* __restrict__ K,
                            _Float16* __restrict__ qh,
                            _Float16* __restrict__ kh) {
    int wv = blockIdx.x * (blockDim.x >> 5) + (threadIdx.x >> 5);
    int lane = threadIdx.x & 31;
    int n = wv >> 3, h = wv & 7;
    int off = n * C_ + h * DH_ + lane;
    float xq = Q[off], xk = K[off];
    float sq = xq * xq, sk = xk * xk;
#pragma unroll
    for (int m = 16; m > 0; m >>= 1) {
        sq += __shfl_xor(sq, m, 32);
        sk += __shfl_xor(sk, m, 32);
    }
    qh[off] = (_Float16)(xq * rsqrtf(sq));
    kh[off] = (_Float16)(xk * rsqrtf(sk));
}

// ---------------------------------------------------------------------------
// Kernel 5: transpose V(N,C) -> vT(C,N) as f16 (K-contiguous B operand for AV).
__global__ void vt_kernel(const float* __restrict__ V, _Float16* __restrict__ vT) {
    int idx = blockIdx.x * blockDim.x + threadIdx.x;
    if (idx >= C_ * N_) return;
    int c = idx >> 10, m = idx & (N_ - 1);
    vT[idx] = (_Float16)V[m * C_ + c];
}

// ---------------------------------------------------------------------------
// Kernel 6: S[h][n][m] = q_n . k_m (head h). One wave per 16x16 tile, K=32.
__global__ void scores_kernel(const _Float16* __restrict__ qh,
                              const _Float16* __restrict__ kh,
                              float* __restrict__ S) {
    int lane = threadIdx.x & 31;
    int ln = lane & 15, hi = lane >> 4;
    int nb = blockIdx.x * 16, mb = blockIdx.y * 16, h = blockIdx.z;
    v16h a = load_a_frag(qh + nb * C_ + h * DH_, C_, lane);
    v16h b = load_b_frag(kh + mb * C_ + h * DH_, C_, lane);
    v8f acc = {};
    acc = __builtin_amdgcn_wmma_f32_16x16x32_f16(false, a, false, b,
                                                 (short)0, acc, false, false);
    float* out = S + (size_t)h * N_ * N_;
#pragma unroll
    for (int r = 0; r < 8; ++r)
        out[(size_t)(nb + r + hi * 8) * N_ + mb + ln] = acc[r];
}

// ---------------------------------------------------------------------------
// Kernel 7: row softmax of S (per head h, row n, over 1024 cols), in place.
__global__ void softmax_kernel(float* __restrict__ S) {
    int n = blockIdx.x, h = blockIdx.y, t = threadIdx.x;
    float* row = S + (size_t)h * N_ * N_ + (size_t)n * N_;
    __shared__ float red[256];
    float v[4]; float mx = -3.4e38f;
#pragma unroll
    for (int i = 0; i < 4; ++i) { v[i] = row[t + i * 256]; mx = fmaxf(mx, v[i]); }
    red[t] = mx; __syncthreads();
    for (int s = 128; s > 0; s >>= 1) {
        if (t < s) red[t] = fmaxf(red[t], red[t + s]);
        __syncthreads();
    }
    mx = red[0]; __syncthreads();
    float sum = 0.f;
#pragma unroll
    for (int i = 0; i < 4; ++i) { v[i] = __expf(v[i] - mx); sum += v[i]; }
    red[t] = sum; __syncthreads();
    for (int s = 128; s > 0; s >>= 1) {
        if (t < s) red[t] += red[t + s];
        __syncthreads();
    }
    float inv = 1.f / red[0];
#pragma unroll
    for (int i = 0; i < 4; ++i) row[t + i * 256] = v[i] * inv;
}

// ---------------------------------------------------------------------------
// Kernel 8: fused position-embedding + log + add-to-softmax, emits f16 attn.
// One thread per (n,m); 8 head accumulators held in registers.
__global__ void pe_kernel(const float* __restrict__ boxes,   // (N,4)
                          const float* __restrict__ conv_w,  // (8,64)
                          const float* __restrict__ conv_b,  // (8)
                          const float* __restrict__ S,
                          _Float16* __restrict__ Ah) {
    __shared__ float bn[16][4], bm[16][4], cw[512], cb[8];
    int tx = threadIdx.x, ty = threadIdx.y;
    int tid = ty * 16 + tx;
    if (tid < 64)       bn[tid >> 2][tid & 3] = boxes[(blockIdx.y * 16 + (tid >> 2)) * 4 + (tid & 3)];
    else if (tid < 128) { int q = tid - 64; bm[q >> 2][q & 3] = boxes[(blockIdx.x * 16 + (q >> 2)) * 4 + (q & 3)]; }
    for (int i = tid; i < 512; i += 256) cw[i] = conv_w[i];
    if (tid < 8) cb[tid] = conv_b[tid];
    __syncthreads();

    int n = blockIdx.y * 16 + ty, m = blockIdx.x * 16 + tx;
    float x1n = bn[ty][0], y1n = bn[ty][1], x2n = bn[ty][2], y2n = bn[ty][3];
    float x1m = bm[tx][0], y1m = bm[tx][1], x2m = bm[tx][2], y2m = bm[tx][3];
    float wn = x2n - x1n + 1.f, hn = y2n - y1n + 1.f;
    float wm = x2m - x1m + 1.f, hm = y2m - y1m + 1.f;
    float cxn = 0.5f * (x1n + x2n), cyn = 0.5f * (y1n + y2n);
    float cxm = 0.5f * (x1m + x2m), cym = 0.5f * (y1m + y2m);
    float pm[4];
    pm[0] = __logf(fabsf((cxn - cxm) / wn) + 0.001f);
    pm[1] = __logf(fabsf((cyn - cym) / hn) + 0.001f);
    pm[2] = __logf(wn / wm);
    pm[3] = __logf(hn / hm);
    // 100 / 1000^(j/8)
    const float fr[8] = {100.0f, 42.169650f, 17.782795f, 7.4989420f,
                         3.1622777f, 1.3335214f, 0.56234133f, 0.23713737f};
    float acc[8];
#pragma unroll
    for (int h = 0; h < 8; ++h) acc[h] = cb[h];
#pragma unroll
    for (int f = 0; f < 4; ++f) {
#pragma unroll
        for (int j = 0; j < 8; ++j) {
            float d = pm[f] * fr[j];
            float sv = __sinf(d), cv = __cosf(d);
#pragma unroll
            for (int h = 0; h < 8; ++h)
                acc[h] += cw[h * 64 + f * 16 + j] * sv + cw[h * 64 + f * 16 + 8 + j] * cv;
        }
    }
#pragma unroll
    for (int h = 0; h < 8; ++h) {
        float pe = fmaxf(acc[h], 0.f);
        float lp = __logf(pe + 1e-6f);
        size_t off = (size_t)h * N_ * N_ + (size_t)n * N_ + m;
        Ah[off] = (_Float16)(S[off] + lp);
    }
}

// ---------------------------------------------------------------------------
// Kernel 9: X2[n][h*32+d] = sum_m A[h][n][m] * v[m][h*32+d]  (WMMA, K loop).
__global__ void av_kernel(const _Float16* __restrict__ Ah,
                          const _Float16* __restrict__ vT,
                          float* __restrict__ X2) {
    int lane = threadIdx.x & 31;
    int ln = lane & 15, hi = lane >> 4;
    int nb = blockIdx.x * 16, db = blockIdx.y * 16, h = blockIdx.z;
    const _Float16* Abase = Ah + (size_t)h * N_ * N_ + (size_t)nb * N_;
    const _Float16* Bbase = vT + (size_t)(h * DH_ + db) * N_;
    v8f acc = {};
    for (int k0 = 0; k0 < N_; k0 += 32) {
        v16h a = load_a_frag(Abase + k0, N_, lane);
        v16h b = load_b_frag(Bbase + k0, N_, lane);
        acc = __builtin_amdgcn_wmma_f32_16x16x32_f16(false, a, false, b,
                                                     (short)0, acc, false, false);
    }
#pragma unroll
    for (int r = 0; r < 8; ++r)
        X2[(nb + r + hi * 8) * C_ + h * DH_ + db + ln] = acc[r];
}

// ---------------------------------------------------------------------------
// Kernel 10: residual + LayerNorm over C=256.
__global__ void ln_kernel(const float* __restrict__ tgt,
                          const float* __restrict__ X2,
                          const float* __restrict__ g,
                          const float* __restrict__ b,
                          float* __restrict__ out) {
    int n = blockIdx.x, t = threadIdx.x;
    __shared__ float red[256];
    float y = tgt[n * C_ + t] + X2[n * C_ + t];
    red[t] = y; __syncthreads();
    for (int s = 128; s > 0; s >>= 1) {
        if (t < s) red[t] += red[t + s];
        __syncthreads();
    }
    float mu = red[0] * (1.f / C_); __syncthreads();
    float d = y - mu;
    red[t] = d * d; __syncthreads();
    for (int s = 128; s > 0; s >>= 1) {
        if (t < s) red[t] += red[t + s];
        __syncthreads();
    }
    float var = red[0] * (1.f / C_);
    out[n * C_ + t] = d * rsqrtf(var + 1e-5f) * g[t] + b[t];
}

// ---------------------------------------------------------------------------
extern "C" void kernel_launch(void* const* d_in, const int* in_sizes, int n_in,
                              void* d_out, int out_size, void* d_ws, size_t ws_size,
                              hipStream_t stream) {
    const float* tgt    = (const float*)d_in[0];
    const float* qpos   = (const float*)d_in[1];
    const float* qedge  = (const float*)d_in[2];
    const float* boxes  = (const float*)d_in[3];
    const float* Wq     = (const float*)d_in[4];
    const float* Wk     = (const float*)d_in[5];
    const float* Wv     = (const float*)d_in[6];
    const float* conv_w = (const float*)d_in[7];
    const float* conv_b = (const float*)d_in[8];
    const float* ln_g   = (const float*)d_in[9];
    const float* ln_b   = (const float*)d_in[10];
    const float* se_w1  = (const float*)d_in[11];
    const float* se_w2  = (const float*)d_in[12];
    float* out = (float*)d_out;

    char* ws = (char*)d_ws;
    size_t off = 0;
    auto alloc = [&](size_t bytes) { char* p = ws + off; off += (bytes + 255) & ~(size_t)255; return p; };
    _Float16* tgth = (_Float16*)alloc(N_ * C_ * 2);
    _Float16* qkh  = (_Float16*)alloc(N_ * C_ * 2);
    _Float16* Wqh  = (_Float16*)alloc(C_ * C_ * 2);
    _Float16* Wkh  = (_Float16*)alloc(C_ * C_ * 2);
    _Float16* Wvh  = (_Float16*)alloc(C_ * C_ * 2);
    float*    Q    = (float*)alloc(N_ * C_ * 4);
    float*    K    = (float*)alloc(N_ * C_ * 4);
    float*    V    = (float*)alloc(N_ * C_ * 4);
    _Float16* qh   = (_Float16*)alloc(N_ * C_ * 2);
    _Float16* kh   = (_Float16*)alloc(N_ * C_ * 2);
    _Float16* vT   = (_Float16*)alloc(C_ * N_ * 2);
    float*    S    = (float*)alloc((size_t)H_ * N_ * N_ * 4);
    _Float16* Ah   = (_Float16*)alloc((size_t)H_ * N_ * N_ * 2);
    float*    X2   = (float*)alloc(N_ * C_ * 4);

    wconv_kernel<<<(C_ * C_ + 255) / 256, 256, 0, stream>>>(Wq, Wk, Wv, Wqh, Wkh, Wvh);
    se_qk_kernel<<<(N_ * C_ + 255) / 256, 256, 0, stream>>>(tgt, qpos, qedge, se_w1, se_w2, qkh, tgth);
    proj_kernel<<<dim3(N_ / 16, C_ / 16, 3), 32, 0, stream>>>(qkh, tgth, Wqh, Wkh, Wvh, Q, K, V);
    norm_kernel<<<(N_ * H_) / 8, 256, 0, stream>>>(Q, K, qh, kh);
    vt_kernel<<<(C_ * N_ + 255) / 256, 256, 0, stream>>>(V, vT);
    scores_kernel<<<dim3(N_ / 16, N_ / 16, H_), 32, 0, stream>>>(qh, kh, S);
    softmax_kernel<<<dim3(N_, H_), 256, 0, stream>>>(S);
    pe_kernel<<<dim3(N_ / 16, N_ / 16), dim3(16, 16), 0, stream>>>(boxes, conv_w, conv_b, S, Ah);
    av_kernel<<<dim3(N_ / 16, DH_ / 16, H_), 32, 0, stream>>>(Ah, vT, X2);
    ln_kernel<<<N_, 256, 0, stream>>>(tgt, X2, ln_g, ln_b, out);
}
